// SIMBAD_block_38405597561098
// MI455X (gfx1250) — compile-verified, matrix operand order checked by reading, not run on hardware
//
#include <hip/hip_runtime.h>
#include <hip/hip_bf16.h>

// ---------------- problem constants ----------------
#define Bz 16
#define Nn 400
#define Ff 64
#define Tt 12
#define Hh 4
#define HD 16
#define BT (Bz * Tt)           // 192
#define BTN (BT * Nn)          // 76800
static const long long SZ_BTNF = (long long)BT * Nn * Ff;   // 4,915,200
static const long long SZ_BNN  = (long long)Bz * Nn * Nn;   // 2,560,000

typedef _Float16 v16h __attribute__((ext_vector_type(16)));
typedef float    v8f  __attribute__((ext_vector_type(8)));

__device__ __forceinline__ v8f wmma_f16(v16h a, v16h b, v8f c) {
  return __builtin_amdgcn_wmma_f32_16x16x32_f16(false, a, false, b, (short)0, c, false, false);
}

// 32 bytes of f16 from LDS as two 16B vector loads (p must be 16B aligned)
__device__ __forceinline__ v16h ldsv16(const _Float16* p) {
  v16h r;
  ((uint4*)&r)[0] = ((const uint4*)p)[0];
  ((uint4*)&r)[1] = ((const uint4*)p)[1];
  return r;
}

// 16 consecutive f32 from global (16B aligned) -> f16 fragment
__device__ __forceinline__ v16h cvt16(const float* p) {
  const float4* q = (const float4*)p;
  float4 a = q[0], b = q[1], c = q[2], d = q[3];
  v16h r;
  r[0]=(_Float16)a.x;  r[1]=(_Float16)a.y;  r[2]=(_Float16)a.z;  r[3]=(_Float16)a.w;
  r[4]=(_Float16)b.x;  r[5]=(_Float16)b.y;  r[6]=(_Float16)b.z;  r[7]=(_Float16)b.w;
  r[8]=(_Float16)c.x;  r[9]=(_Float16)c.y;  r[10]=(_Float16)c.z; r[11]=(_Float16)c.w;
  r[12]=(_Float16)d.x; r[13]=(_Float16)d.y; r[14]=(_Float16)d.z; r[15]=(_Float16)d.w;
  return r;
}

// =====================================================================
// Generic batched WMMA GEMM: C[z] = A[z/adiv] (MxK) @ B[z/bdiv] (KxNc)
// 128 threads = 4 waves; workgroup tile 64x64; wave tile 16x64.
// EPI 0: C=acc   EPI 1: C=acc+bias[col]   EPI 2: C=sigmoid(acc+addin[row,col])
// GUARD=false: M%64==0, Nc==64, K%32==0 guaranteed -> zero bounds checks.
// GUARD=true : K%16==0 still assumed -> each 16-elem group all/none valid.
// =====================================================================
template <int EPI, bool GUARD>
__global__ __launch_bounds__(128) void k_gemm(
    const float* __restrict__ A, long long sA, int adiv,
    const float* __restrict__ Bm, long long sB, int bdiv,
    float* __restrict__ C, long long sC,
    const float* __restrict__ bias,
    const float* __restrict__ addin, long long sAdd,
    int M, int Nc, int K, int lda, int ldb, int ldc)
{
  const int z = blockIdx.z;
  const float* Ab = A + (long long)(z / adiv) * sA;
  const float* Bb = Bm + (long long)(z / bdiv) * sB;
  float* Cb = C + (long long)z * sC;
  const float* Addb = (EPI == 2) ? (addin + (long long)z * sAdd) : nullptr;

  const int rowBase = blockIdx.x * 64;
  const int colBase = blockIdx.y * 64;
  const int tid  = threadIdx.x;
  const int wave = tid >> 5;
  const int lane = tid & 31;
  const int lh   = lane >> 4;   // 0/1 : k-half selector
  const int ll   = lane & 15;

  __shared__ __align__(32) _Float16 Bt[64 * 40];   // [col][k] chunk, stride 40 halfs

  const int myRow0 = rowBase + wave * 16;
  v8f acc0 = {}, acc1 = {}, acc2 = {}, acc3 = {};

  for (int k0 = 0; k0 < K; k0 += 32) {
    __syncthreads();
    {  // cooperative stage of 32-K chunk of B, transposed + f16; one branch per strip
      const int c   = tid & 63;
      const int kk0 = (tid >> 6) * 16;
      const int col = colBase + c;
      _Float16* dst = &Bt[c * 40 + kk0];
      if (!GUARD || ((col < Nc) & (k0 + kk0 + 16 <= K))) {
        const float* bp = Bb + (long long)(k0 + kk0) * ldb + col;
#pragma unroll
        for (int i = 0; i < 16; ++i) dst[i] = (_Float16)bp[(long long)i * ldb];
      } else {
#pragma unroll
        for (int i = 0; i < 16; ++i) dst[i] = (_Float16)0.0f;
      }
    }
    __syncthreads();
    // A fragment 16x32 (lane ll = row, lh selects K half); one branch per fragment
    v16h af;
    {
      const int row = myRow0 + ll;
      const int kb  = k0 + lh * 16;
      if (!GUARD || ((row < M) & (kb + 16 <= K))) {
        af = cvt16(Ab + (long long)row * lda + kb);
      } else {
#pragma unroll
        for (int i = 0; i < 16; ++i) af[i] = (_Float16)0.0f;
      }
    }
#pragma unroll
    for (int ct = 0; ct < 4; ++ct) {
      v16h bf = ldsv16(&Bt[(ct * 16 + ll) * 40 + lh * 16]);
      v8f& acc = (ct == 0) ? acc0 : (ct == 1) ? acc1 : (ct == 2) ? acc2 : acc3;
      acc = wmma_f16(af, bf, acc);
    }
  }

#pragma unroll
  for (int ct = 0; ct < 4; ++ct) {
    const v8f& acc = (ct == 0) ? acc0 : (ct == 1) ? acc1 : (ct == 2) ? acc2 : acc3;
    const int col = colBase + ct * 16 + ll;
    if (!GUARD || col < Nc) {
#pragma unroll
      for (int r = 0; r < 8; ++r) {
        const int row = myRow0 + lh * 8 + r;   // lanes 0-15: M=r, lanes 16-31: M=8+r
        if (!GUARD || row < M) {
          float v = acc[r];
          if (EPI == 1) v += bias[col];
          if (EPI == 2) v = 1.0f / (1.0f + __expf(-(v + Addb[(long long)row * ldc + col])));
          Cb[(long long)row * ldc + col] = v;
        }
      }
    }
  }
}

// =====================================================================
// Spatial attention: one workgroup per (b*T+t, h). 4 waves, wave32.
// Q/K staged in LDS zero-padded to d=32 (so the K=16..31 fragment half
// reads real zeros -> fully unconditional b128 LDS loads). V transposed.
// =====================================================================
#define QKROW 32
#define VROW 416

__global__ __launch_bounds__(128) void k_attn_spatial(
    const float* __restrict__ Q, const float* __restrict__ Kg,
    const float* __restrict__ V, float* __restrict__ O)
{
  extern __shared__ __align__(32) char smem[];
  _Float16* Qsh = (_Float16*)smem;                      // [400][32], d>=16 zero
  _Float16* Ksh = Qsh + Nn * QKROW;                     // [400][32], d>=16 zero
  _Float16* Vsh = Ksh + Nn * QKROW;                     // [16][VROW] (transposed)
  float*    Ssh = (float*)(Vsh + HD * VROW);            // 4 waves x [16][416]
  float*    Sum = Ssh + 4 * 16 * 416;                   // 4 waves x 16 inv-sums

  const int bt = blockIdx.x;
  const int h  = blockIdx.y;
  const int tid = threadIdx.x;
  const int wave = tid >> 5, lane = tid & 31, lh = lane >> 4, ll = lane & 15;
  const long long base = (long long)bt * Nn * Ff + h * HD;

  for (int e = tid; e < Nn * HD; e += 128) {
    const int n = e >> 4, d = e & 15;
    Qsh[n * QKROW + d]      = (_Float16)(Q[base + (long long)n * Ff + d] * 0.25f); // 1/sqrt(HD)
    Qsh[n * QKROW + 16 + d] = (_Float16)0.0f;
    Ksh[n * QKROW + d]      = (_Float16)Kg[base + (long long)n * Ff + d];
    Ksh[n * QKROW + 16 + d] = (_Float16)0.0f;
    Vsh[d * VROW + n]       = (_Float16)V[base + (long long)n * Ff + d];
  }
  for (int e = tid; e < HD * 16; e += 128)              // zero-pad V keys 400..415
    Vsh[(e >> 4) * VROW + 400 + (e & 15)] = (_Float16)0.0f;
  float* Sw = Ssh + wave * 16 * 416;
  for (int e = lane; e < 16 * 16; e += 32)              // zero-pad P cols 400..415
    Sw[(e >> 4) * 416 + 400 + (e & 15)] = 0.0f;
  __syncthreads();

  float* sumw = Sum + wave * 16;

  for (int qt = wave; qt < 25; qt += 4) {               // wave-uniform
    const int qBase = qt * 16;
    const v16h qf = ldsv16(&Qsh[(qBase + ll) * QKROW + lh * 16]);
    for (int kt = 0; kt < 25; ++kt) {                   // S = Q K^T
      const v16h kf = ldsv16(&Ksh[(kt * 16 + ll) * QKROW + lh * 16]);
      v8f s = {};
      s = wmma_f16(qf, kf, s);
#pragma unroll
      for (int r = 0; r < 8; ++r)
        Sw[(lh * 8 + r) * 416 + kt * 16 + ll] = s[r];
    }
    asm volatile("s_wait_dscnt 0" ::: "memory");        // wave-local LDS visibility
    for (int r = 0; r < 16; ++r) {                      // softmax rows
      float* row = Sw + r * 416;
      float m = -1e30f;
      for (int c = lane; c < Nn; c += 32) m = fmaxf(m, row[c]);
#pragma unroll
      for (int o = 16; o > 0; o >>= 1) m = fmaxf(m, __shfl_xor(m, o, 32));
      float sum = 0.0f;
      for (int c = lane; c < Nn; c += 32) {
        float e = __expf(row[c] - m);
        row[c] = e;
        sum += e;
      }
#pragma unroll
      for (int o = 16; o > 0; o >>= 1) sum += __shfl_xor(sum, o, 32);
      if (lane == 0) sumw[r] = 1.0f / sum;
    }
    asm volatile("s_wait_dscnt 0" ::: "memory");
    v8f o = {};                                         // O = P V  (13 k-chunks of 32)
    for (int kc = 0; kc < 13; ++kc) {
      const int k0 = kc * 32;
      const v16h pf = cvt16(Sw + ll * 416 + k0 + lh * 16);
      const v16h vf = ldsv16(&Vsh[ll * VROW + k0 + lh * 16]);
      o = wmma_f16(pf, vf, o);
    }
#pragma unroll
    for (int r = 0; r < 8; ++r) {
      const int row = qBase + lh * 8 + r;
      O[base + (long long)row * Ff + ll] = o[r] * sumw[lh * 8 + r];
    }
  }
}

// ---------------- temporal attention: one WG per (b,n), L=12 ----------------
__global__ __launch_bounds__(64) void k_attn_temporal(
    const float* __restrict__ Q, const float* __restrict__ Kg,
    const float* __restrict__ V, float* __restrict__ O)
{
  __shared__ float q[Tt * Ff], k[Tt * Ff], v[Tt * Ff];
  const long long base = (long long)blockIdx.x * Tt * Ff;
  for (int e = threadIdx.x; e < Tt * Ff; e += 64) {
    q[e] = Q[base + e]; k[e] = Kg[base + e]; v[e] = V[base + e];
  }
  __syncthreads();
  const int tid = threadIdx.x;
  if (tid < Tt * Hh) {
    const int t = tid >> 2, h = tid & 3;
    const float* qr = q + t * Ff + h * HD;
    float s[Tt], m = -1e30f;
    for (int j = 0; j < Tt; ++j) {
      const float* kr = k + j * Ff + h * HD;
      float a = 0.0f;
      for (int d = 0; d < HD; ++d) a += qr[d] * kr[d];
      s[j] = a * 0.25f;
      m = fmaxf(m, s[j]);
    }
    float sum = 0.0f;
    for (int j = 0; j < Tt; ++j) { s[j] = __expf(s[j] - m); sum += s[j]; }
    const float inv = 1.0f / sum;
    for (int d = 0; d < HD; ++d) {
      float a = 0.0f;
      for (int j = 0; j < Tt; ++j) a += s[j] * v[j * Ff + h * HD + d];
      O[base + t * Ff + h * HD + d] = a * inv;
    }
  }
}

// ---------------- layout shuffles ----------------
__global__ void k_xt(const float* __restrict__ x, float* __restrict__ XT) {
  long long i = (long long)blockIdx.x * blockDim.x + threadIdx.x;
  if (i >= SZ_BTNF) return;                       // (b,t,n,f) <- x[b,n,f,t]
  int f = i & 63; long long r = i >> 6;
  int n = (int)(r % Nn); r /= Nn;
  int t = (int)(r % Tt); int b = (int)(r / Tt);
  XT[i] = x[(((long long)b * Nn + n) * Ff + f) * Tt + t];
}
__global__ void k_xt2(const float* __restrict__ x, float* __restrict__ XT2) {
  long long i = (long long)blockIdx.x * blockDim.x + threadIdx.x;
  if (i >= SZ_BTNF) return;                       // (b,n,t,f) <- x[b,n,f,t]
  int f = i & 63; long long r = i >> 6;
  int t = (int)(r % Tt); r /= Tt;
  int n = (int)(r % Nn); int b = (int)(r / Nn);
  XT2[i] = x[(((long long)b * Nn + n) * Ff + f) * Tt + t];
}

// ---------------- adjacency prep ----------------
__global__ void k_ncount(const float* __restrict__ adj, float* __restrict__ nc) {
  int i = blockIdx.x * blockDim.x + threadIdx.x;
  if (i >= Nn) return;
  float s = 0.0f;
  for (int j = 0; j < Nn; ++j) s += adj[i * Nn + j];
  nc[i] = s - 1.0f + 1e-10f;                      // (adj - eye).sum(axis=1)
}

__global__ void k_adjw(const float* __restrict__ adj, const float* __restrict__ nc,
                       const float* __restrict__ wadj, const float* __restrict__ wadj2,
                       const float* __restrict__ th1, const float* __restrict__ th2,
                       const float* __restrict__ th3, const float* __restrict__ th4,
                       float* __restrict__ A1a, float* __restrict__ A2a,
                       float* __restrict__ A1b, float* __restrict__ A2b) {
  long long idx = (long long)blockIdx.x * blockDim.x + threadIdx.x;
  if (idx >= SZ_BNN) return;
  int j = (int)(idx % Nn); long long r = idx / Nn;
  int i = (int)(r % Nn);
  float a = adj[i * Nn + j];
  float w1 = 0.f, w2 = 0.f, w3 = 0.f, w4 = 0.f;
  if (i == j) { w1 = th1[i]; w2 = th2[i]; w3 = th3[i]; w4 = th4[i]; }
  else if (a > 0.f) {
    float inv = 1.0f / nc[i];
    w1 = th1[i] * inv; w2 = th2[i] * inv; w3 = th3[i] * inv; w4 = th4[i] * inv;
  }
  float wv = wadj[idx], wv2 = wadj2[idx];
  A1a[idx] = wv * w1;  A2a[idx] = wv2 * w2;
  A1b[idx] = wv * w3;  A2b[idx] = wv2 * w4;
}

// ---------------- segment max over graph neighbors ----------------
__global__ __launch_bounds__(64) void k_maxagg(const float* __restrict__ h,
                                               const float* __restrict__ adj,
                                               float* __restrict__ out) {
  __shared__ float arow[Nn];
  const int n = blockIdx.x % Nn;
  const long long bt = blockIdx.x / Nn;
  for (int e = threadIdx.x; e < Nn; e += 64) arow[e] = adj[n * Nn + e];
  __syncthreads();
  const int f = threadIdx.x;
  const float* hb = h + bt * Nn * Ff;
  float m = -1e30f;
  for (int s = 0; s < Nn; ++s)
    if (arow[s] > 0.f) m = fmaxf(m, hb[(long long)s * Ff + f]);
  out[(bt * Nn + n) * Ff + f] = m;
}

// ---------------- gated temporal conv (kernel 3, pad 1 along T) ----------------
__global__ __launch_bounds__(256) void k_gated(const float* __restrict__ TA,
                                               const float* __restrict__ tw, const float* __restrict__ tb,
                                               const float* __restrict__ gw, const float* __restrict__ gb,
                                               float* __restrict__ out) {
  __shared__ float xs[Tt * Ff];
  const long long bn = blockIdx.x;
  const float* src = TA + bn * Tt * Ff;
  for (int e = threadIdx.x; e < Tt * Ff; e += 256) xs[e] = src[e];
  __syncthreads();
  for (int e = threadIdx.x; e < Tt * Ff; e += 256) {
    const int o = e & 63, t = e >> 6;
    float at = tb[o], ag = gb[o];
    for (int kt = 0; kt < 3; ++kt) {
      const int tt = t + kt - 1;
      if (tt < 0 || tt >= Tt) continue;
      const float* xr = xs + tt * Ff;
      const float* twr = tw + o * (Ff * 3) + kt;   // tc_w[o][i][0][kt]
      const float* gwr = gw + o * (Ff * 3) + kt;
      for (int i = 0; i < Ff; ++i) {
        const float v = xr[i];
        at += v * twr[i * 3];
        ag += v * gwr[i * 3];
      }
    }
    out[bn * Tt * Ff + e] = at * (1.0f / (1.0f + __expf(-ag)));
  }
}

// ---------------- mask + residual + ReLU + LayerNorm(F) -> output layout ----------------
__global__ __launch_bounds__(64) void k_combine(const float* __restrict__ M2,
                                                const float* __restrict__ GT,
                                                const float* __restrict__ x,
                                                const float* __restrict__ mw,
                                                const float* __restrict__ md,
                                                const float* __restrict__ lg,
                                                const float* __restrict__ lb,
                                                const int* __restrict__ maskp,
                                                float* __restrict__ out) {
  __shared__ float red[4];
  const long long id = blockIdx.x;                // (b*T+t)*N + n
  const int n = (int)(id % Nn); long long r = id / Nn;
  const int t = (int)(r % Tt); const int b = (int)(r / Tt);
  const int f = threadIdx.x;
  float last = M2[id * Ff + f];
  if (maskp[0]) last *= mw[b * Nn + n] * md[b * Nn + n];
  const float g  = GT[(((long long)b * Nn + n) * Tt + t) * Ff + f];
  const float xv = x[(((long long)b * Nn + n) * Ff + f) * Tt + t];
  const float rv = fmaxf(last + g + xv, 0.0f);
  float s = rv, s2 = rv * rv;
#pragma unroll
  for (int o = 16; o > 0; o >>= 1) { s += __shfl_xor(s, o, 32); s2 += __shfl_xor(s2, o, 32); }
  const int wv = threadIdx.x >> 5;
  if ((threadIdx.x & 31) == 0) { red[wv * 2] = s; red[wv * 2 + 1] = s2; }
  __syncthreads();
  const float S = red[0] + red[2], S2 = red[1] + red[3];
  const float mu  = S * (1.0f / Ff);
  const float var = S2 * (1.0f / Ff) - mu * mu;
  const float nrm = (rv - mu) * rsqrtf(var + 1e-5f);
  out[(((long long)b * Nn + n) * Ff + f) * Tt + t] = nrm * lg[f] + lb[f];
}

// =====================================================================
extern "C" void kernel_launch(void* const* d_in, const int* in_sizes, int n_in,
                              void* d_out, int out_size, void* d_ws, size_t ws_size,
                              hipStream_t stream) {
  (void)in_sizes; (void)n_in; (void)out_size; (void)ws_size;
  const float* x    = (const float*)d_in[0];
  const float* wadj = (const float*)d_in[1];
  const float* mw   = (const float*)d_in[2];
  const float* md   = (const float*)d_in[3];
  const float* th1  = (const float*)d_in[4];
  const float* th2  = (const float*)d_in[5];
  const float* th3  = (const float*)d_in[6];
  const float* th4  = (const float*)d_in[7];
  const float* gw1a = (const float*)d_in[8];
  const float* gw1b = (const float*)d_in[9];
  const float* gw2a = (const float*)d_in[10];
  const float* gw2b = (const float*)d_in[11];
  const float* mp_w = (const float*)d_in[12];
  const float* mp_b = (const float*)d_in[13];
  const float* tc_w = (const float*)d_in[14];
  const float* tc_b = (const float*)d_in[15];
  const float* gc_w = (const float*)d_in[16];
  const float* gc_b = (const float*)d_in[17];
  const float* sq_w = (const float*)d_in[18];
  const float* sq_b = (const float*)d_in[19];
  const float* sk_w = (const float*)d_in[20];
  const float* sk_b = (const float*)d_in[21];
  const float* sv_w = (const float*)d_in[22];
  const float* sv_b = (const float*)d_in[23];
  const float* so_w = (const float*)d_in[24];
  const float* so_b = (const float*)d_in[25];
  const float* tq_w = (const float*)d_in[26];
  const float* tq_b = (const float*)d_in[27];
  const float* tk_w = (const float*)d_in[28];
  const float* tk_b = (const float*)d_in[29];
  const float* tv_w = (const float*)d_in[30];
  const float* tv_b = (const float*)d_in[31];
  const float* to_w = (const float*)d_in[32];
  const float* to_b = (const float*)d_in[33];
  const float* ln_g = (const float*)d_in[34];
  const float* ln_b = (const float*)d_in[35];
  const float* adj  = (const float*)d_in[36];
  const int* maskp  = (const int*)d_in[38];
  float* out = (float*)d_out;

  float* ws = (float*)d_ws;
  float* S0 = ws + 0 * SZ_BTNF;   // XT -> XT2 -> gated
  float* S1 = ws + 1 * SZ_BTNF;   // Qs -> Qt -> f1 -> f1b
  float* S2 = ws + 2 * SZ_BTNF;   // Ks -> Kt -> f2 -> f2b
  float* S3 = ws + 3 * SZ_BTNF;   // Vs -> Vt -> tmp
  float* S4 = ws + 4 * SZ_BTNF;   // AO -> AO2 -> mean1 -> mean2
  float* S5 = ws + 5 * SZ_BTNF;   // SA -> h
  float* S6 = ws + 6 * SZ_BTNF;   // TA
  float* S7 = ws + 7 * SZ_BTNF;   // maxp
  float* A1a = ws + 8 * SZ_BTNF;
  float* A2a = A1a + SZ_BNN;
  float* A1b = A2a + SZ_BNN;
  float* A2b = A1b + SZ_BNN;
  float* W2  = A2b + SZ_BNN;      // wadj @ wadj
  float* NC  = W2 + SZ_BNN;       // ncount[N]

  const dim3 b128(128), b256(256), b64(64);
  const dim3 gProj(BTN / 64, 1, 1);                 // 1200 row tiles, M=76800
  const dim3 gAgg(7, 1, BT);                        // M=400 row tiles, per (b,t)
  const int gEl = (int)((SZ_BTNF + 255) / 256);
  const int gAd = (int)((SZ_BNN + 255) / 256);

  k_ncount<<<dim3(7), b64, 0, stream>>>(adj, NC);

  // ---- spatial attention path (layout B,T,N,F) ----
  k_xt<<<dim3(gEl), b256, 0, stream>>>(x, S0);
  k_gemm<1,false><<<gProj, b128, 0, stream>>>(S0, 0, 1, sq_w, 0, 1, S1, 0, sq_b, nullptr, 0, BTN, Ff, Ff, Ff, Ff, Ff);
  k_gemm<1,false><<<gProj, b128, 0, stream>>>(S0, 0, 1, sk_w, 0, 1, S2, 0, sk_b, nullptr, 0, BTN, Ff, Ff, Ff, Ff, Ff);
  k_gemm<1,false><<<gProj, b128, 0, stream>>>(S0, 0, 1, sv_w, 0, 1, S3, 0, sv_b, nullptr, 0, BTN, Ff, Ff, Ff, Ff, Ff);
  const size_t attnLds = (size_t)Nn * QKROW * 2 * 2 + (size_t)HD * VROW * 2
                       + (size_t)4 * 16 * 416 * 4 + 4 * 16 * 4;   // ~171 KB < 320 KB/WG
  k_attn_spatial<<<dim3(BT, Hh), b128, attnLds, stream>>>(S1, S2, S3, S4);
  k_gemm<1,false><<<gProj, b128, 0, stream>>>(S4, 0, 1, so_w, 0, 1, S5, 0, so_b, nullptr, 0, BTN, Ff, Ff, Ff, Ff, Ff);

  // ---- temporal attention path (layout B,N,T,F) ----
  k_xt2<<<dim3(gEl), b256, 0, stream>>>(x, S0);
  k_gemm<1,false><<<gProj, b128, 0, stream>>>(S0, 0, 1, tq_w, 0, 1, S1, 0, tq_b, nullptr, 0, BTN, Ff, Ff, Ff, Ff, Ff);
  k_gemm<1,false><<<gProj, b128, 0, stream>>>(S0, 0, 1, tk_w, 0, 1, S2, 0, tk_b, nullptr, 0, BTN, Ff, Ff, Ff, Ff, Ff);
  k_gemm<1,false><<<gProj, b128, 0, stream>>>(S0, 0, 1, tv_w, 0, 1, S3, 0, tv_b, nullptr, 0, BTN, Ff, Ff, Ff, Ff, Ff);
  k_attn_temporal<<<dim3(Bz * Nn), b64, 0, stream>>>(S1, S2, S3, S4);
  k_gemm<1,false><<<gProj, b128, 0, stream>>>(S4, 0, 1, to_w, 0, 1, S6, 0, to_b, nullptr, 0, BTN, Ff, Ff, Ff, Ff, Ff);

  // ---- graph weights: wadj@wadj and the 4 weighted adjacency matrices ----
  k_gemm<0,true><<<dim3(7, 7, Bz), b128, 0, stream>>>(wadj, (long long)Nn * Nn, 1, wadj, (long long)Nn * Nn, 1,
                                                      W2, (long long)Nn * Nn, nullptr, nullptr, 0,
                                                      Nn, Nn, Nn, Nn, Nn, Nn);
  k_adjw<<<dim3(gAd), b256, 0, stream>>>(adj, NC, wadj, W2, th1, th2, th3, th4, A1a, A2a, A1b, A2b);

  // ---- agg_conv #1: mean1 = sigmoid(A1a@SA @ gw1a + A2a@SA @ gw1b) ----
  k_gemm<0,true><<<gAgg, b128, 0, stream>>>(A1a, (long long)Nn * Nn, Tt, S5, (long long)Nn * Ff, 1,
                                            S1, (long long)Nn * Ff, nullptr, nullptr, 0, Nn, Ff, Nn, Nn, Ff, Ff);
  k_gemm<0,true><<<gAgg, b128, 0, stream>>>(A2a, (long long)Nn * Nn, Tt, S5, (long long)Nn * Ff, 1,
                                            S2, (long long)Nn * Ff, nullptr, nullptr, 0, Nn, Ff, Nn, Nn, Ff, Ff);
  k_gemm<0,false><<<gProj, b128, 0, stream>>>(S2, 0, 1, gw1b, 0, 1, S3, 0, nullptr, nullptr, 0, BTN, Ff, Ff, Ff, Ff, Ff);
  k_gemm<2,false><<<gProj, b128, 0, stream>>>(S1, 0, 1, gw1a, 0, 1, S4, 0, nullptr, S3, 0, BTN, Ff, Ff, Ff, Ff, Ff);

  // ---- max aggregation ----
  k_gemm<1,false><<<gProj, b128, 0, stream>>>(S4, 0, 1, mp_w, 0, 1, S5, 0, mp_b, nullptr, 0, BTN, Ff, Ff, Ff, Ff, Ff);
  k_maxagg<<<dim3(BTN), b64, 0, stream>>>(S5, adj, S7);

  // ---- agg_conv #2: mean2 ----
  k_gemm<0,true><<<gAgg, b128, 0, stream>>>(A1b, (long long)Nn * Nn, Tt, S7, (long long)Nn * Ff, 1,
                                            S1, (long long)Nn * Ff, nullptr, nullptr, 0, Nn, Ff, Nn, Nn, Ff, Ff);
  k_gemm<0,true><<<gAgg, b128, 0, stream>>>(A2b, (long long)Nn * Nn, Tt, S7, (long long)Nn * Ff, 1,
                                            S2, (long long)Nn * Ff, nullptr, nullptr, 0, Nn, Ff, Nn, Nn, Ff, Ff);
  k_gemm<0,false><<<gProj, b128, 0, stream>>>(S2, 0, 1, gw2b, 0, 1, S3, 0, nullptr, nullptr, 0, BTN, Ff, Ff, Ff, Ff, Ff);
  k_gemm<2,false><<<gProj, b128, 0, stream>>>(S1, 0, 1, gw2a, 0, 1, S4, 0, nullptr, S3, 0, BTN, Ff, Ff, Ff, Ff, Ff);

  // ---- gated temporal conv + final fuse ----
  k_gated<<<dim3(Bz * Nn), b256, 0, stream>>>(S6, tc_w, tc_b, gc_w, gc_b, S0);
  k_combine<<<dim3(BTN), b64, 0, stream>>>(S4, S0, x, mw, md, ln_g, ln_b, maskp, out);
}
